// SA_Layer_1451698946594
// MI455X (gfx1250) — matrix operand order repaired and checked by reading, not compiled, not run on hardware
//
#include <hip/hip_runtime.h>
#include <hip/hip_bf16.h>
#include <stdint.h>

#define BATCH 8
#define CCH   256
#define NPTS  4096
#define DQK   64

typedef __bf16 bf16;
typedef float v8f   __attribute__((ext_vector_type(8)));
typedef bf16  v16bf __attribute__((ext_vector_type(16)));
typedef unsigned int u32x4 __attribute__((ext_vector_type(4)));
typedef int i32x8 __attribute__((ext_vector_type(8)));
typedef int i32x4 __attribute__((ext_vector_type(4)));

#if __has_builtin(__builtin_amdgcn_tensor_load_to_lds) && \
    __has_builtin(__builtin_amdgcn_s_wait_tensorcnt)
#define USE_TDM 1
#else
#define USE_TDM 0
#endif

__device__ __forceinline__ v8f wmma_bf16(v16bf a, v16bf b, v8f c) {
  return __builtin_amdgcn_wmma_f32_16x16x32_bf16(false, a, false, b, (short)0, c,
                                                 false, false);
}

__device__ __forceinline__ v8f zero8() {
  v8f z;
#pragma unroll
  for (int i = 0; i < 8; ++i) z[i] = 0.0f;
  return z;
}

// ---- Tensor Data Mover: 2D bf16 tile (tile_w x tile_h rows, row stride
// `stride` elements) global -> LDS row-major; tracked by TENSORcnt. ----
__device__ __forceinline__ void tdm_load_tile_bf16(void* lds_dst, const bf16* gsrc,
                                                   int tile_w, int tile_h, int stride) {
#if USE_TDM
  unsigned long long ga = (unsigned long long)(uintptr_t)gsrc;
  unsigned lds_off = (unsigned)(uintptr_t)lds_dst;  // low 32 bits = LDS byte offset
  u32x4 g0;
  g0[0] = 1u;                                        // count=1, user descriptor
  g0[1] = lds_off;                                   // lds_addr (bytes)
  g0[2] = (unsigned)ga;                              // global_addr[31:0]
  g0[3] = (unsigned)((ga >> 32) & 0x01ffffffu) | 0x80000000u;  // ga[56:32]|type=2
  unsigned td0 = (unsigned)stride;                   // tensor_dim0 (elems)
  unsigned td1 = 1u << 20;                           // tensor_dim1 (big)
  i32x8 g1;
  g1[0] = (int)(1u << 16);                           // data_size=1 (2 bytes)
  g1[1] = (int)((td0 & 0xffffu) << 16);              // abar_addr=0 | td0[15:0]
  g1[2] = (int)((td0 >> 16) | ((td1 & 0xffffu) << 16));
  g1[3] = (int)((td1 >> 16) | ((unsigned)tile_w << 16));  // tile_dim0
  g1[4] = (int)(unsigned)tile_h;                     // tile_dim1 | tile_dim2=0
  g1[5] = (int)(unsigned)stride;                     // dim0_stride[31:0]
  g1[6] = 0;
  g1[7] = 0;
  i32x4 z4;
  z4[0] = z4[1] = z4[2] = z4[3] = 0;
  i32x8 z8;
#pragma unroll
  for (int i = 0; i < 8; ++i) z8[i] = 0;
  __builtin_amdgcn_tensor_load_to_lds(g0, g1, z4, z4, z8, 0);
#else
  (void)lds_dst; (void)gsrc; (void)tile_w; (void)tile_h; (void)stride;
#endif
}

__device__ __forceinline__ void tdm_wait() {
#if USE_TDM
  __builtin_amdgcn_s_wait_tensorcnt(0);
#endif
}

// Manual fallback: 64 rows x 32 bf16 B tile copy (row-major), 128 threads.
__device__ __forceinline__ void copy_b_tile(bf16* bt, const bf16* src, int stride) {
#if !USE_TDM
#pragma unroll
  for (int e = 0; e < 2; ++e) {
    int uid = threadIdx.x + 128 * e;     // 0..255
    int n = uid >> 2, q = uid & 3;
    *(uint4*)&bt[n * 32 + q * 8] = *(const uint4*)(src + (size_t)n * stride + q * 8);
  }
#else
  (void)bt; (void)src; (void)stride;
#endif
}

// A fragment: 16(M) x 32(K) bf16 tile, row-major in LDS, leading dim ld (elems).
__device__ __forceinline__ v16bf frag_a(const bf16* tile, int ld) {
  int lane = threadIdx.x & 31;
  int m  = lane & 15;
  int kb = (lane >> 4) << 3;
  union { unsigned int u[8]; v16bf v; } r;
#pragma unroll
  for (int i = 0; i < 8; ++i) {
    int k0 = kb + ((i & 3) << 1) + ((i >> 2) << 4);
    r.u[i] = *(const unsigned int*)(tile + m * ld + k0);
  }
  return r.v;
}

// B fragment from K-transposed LDS tile [16(N)][ld(K)], K contiguous.
__device__ __forceinline__ v16bf frag_b(const bf16* tileT, int ld) {
  int lane = threadIdx.x & 31;
  int n  = lane & 15;
  int kb = (lane >> 4) << 3;
  union { unsigned int u[8]; v16bf v; } r;
#pragma unroll
  for (int i = 0; i < 8; ++i) {
    int k0 = kb + ((i & 3) << 1) + ((i >> 2) << 4);
    r.u[i] = *(const unsigned int*)(tileT + n * ld + k0);
  }
  return r.v;
}

__device__ __forceinline__ float rmax16(float v) {
#pragma unroll
  for (int o = 8; o >= 1; o >>= 1) v = fmaxf(v, __shfl_xor(v, o, 32));
  return v;
}
__device__ __forceinline__ float rsum16(float v) {
#pragma unroll
  for (int o = 8; o >= 1; o >>= 1) v += __shfl_xor(v, o, 32);
  return v;
}

// ---------------------------------------------------------------------------
// K0: xT[b,n,c] = (bf16)x[b,c,n] — one-time LDS-tiled transpose+convert
// ---------------------------------------------------------------------------
__global__ void __launch_bounds__(256)
xpose_kernel(const float* __restrict__ x, bf16* __restrict__ xT) {
  int b = blockIdx.z, c0 = blockIdx.y * 64, n0 = blockIdx.x * 64;
  int tid = threadIdx.x;
  __shared__ __align__(16) bf16 tile[64 * 64];
  const float* xb = x + ((size_t)b * CCH + c0) * NPTS + n0;
  float4 v[4];
#pragma unroll
  for (int e = 0; e < 4; ++e) {
    int uid = tid + 256 * e;             // 0..1023
    int c = uid >> 4, q = uid & 15;
    v[e] = *(const float4*)(xb + (size_t)c * NPTS + 4 * q);
  }
#pragma unroll
  for (int e = 0; e < 4; ++e) {
    int uid = tid + 256 * e;
    int c = uid >> 4, q = uid & 15;
    const float* f = (const float*)&v[e];
    union { uint2 u2; bf16 h[4]; } pk;
#pragma unroll
    for (int t = 0; t < 4; ++t) pk.h[t] = (bf16)f[t];
    *(uint2*)&tile[c * 64 + 4 * q] = pk.u2;
  }
  __syncthreads();
  bf16* dst = xT + ((size_t)b * NPTS + n0) * CCH + c0;
#pragma unroll
  for (int e = 0; e < 2; ++e) {
    int uid = tid + 256 * e;             // 0..511
    int n = uid >> 3, qc = uid & 7;
    union { uint4 q4; bf16 h[8]; } o;
#pragma unroll
    for (int t = 0; t < 8; ++t) o.h[t] = tile[(qc * 8 + t) * 64 + n];
    *(uint4*)(dst + (size_t)n * CCH + qc * 8) = o.q4;
  }
}

// ---------------------------------------------------------------------------
// K1: proj: out[b,d,n] = sum_c W[d,c]*xT[b,n,c] (+bias); optional outT[b,n,d]
// B tiles come from xT (row-major, stride CCH) -> TDM, no transpose staging.
// ---------------------------------------------------------------------------
__global__ void __launch_bounds__(128)
proj_kernel(const float* __restrict__ W, const float* __restrict__ bias,
            const bf16* __restrict__ xT, bf16* __restrict__ out,
            bf16* __restrict__ outT, int DO) {
  int b = blockIdx.z, d0 = blockIdx.y * 64, n0 = blockIdx.x * 64;
  int wave = threadIdx.x >> 5, lane = threadIdx.x & 31;
  __shared__ __align__(16) bf16 a_lds[4][16 * 32];
  __shared__ __align__(16) bf16 bt_lds[64 * 32];
  v8f acc[4];
#pragma unroll
  for (int j = 0; j < 4; ++j) acc[j] = zero8();

  const float* wrow = W + (size_t)(d0 + wave * 16) * CCH;
  const bf16* xrow = xT + ((size_t)b * NPTS + n0) * CCH;

  for (int k0 = 0; k0 < CCH; k0 += 32) {
    // A: W rows f32 -> bf16
    float4 wa[4];
#pragma unroll
    for (int e = 0; e < 4; ++e) {
      int uid = lane + 32 * e;
      int r = uid >> 3, q = uid & 7;
      wa[e] = *(const float4*)(wrow + (size_t)r * CCH + k0 + 4 * q);
    }
#pragma unroll
    for (int e = 0; e < 4; ++e) {
      int uid = lane + 32 * e;
      int r = uid >> 3, q = uid & 7;
      const float* f = (const float*)&wa[e];
      union { uint2 u2; bf16 h[4]; } pk;
#pragma unroll
      for (int c = 0; c < 4; ++c) pk.h[c] = (bf16)f[c];
      *(uint2*)&a_lds[wave][r * 32 + 4 * q] = pk.u2;
    }
    // B: xT rows (each wave moves its 16-row quarter via TDM)
#if USE_TDM
    tdm_load_tile_bf16(bt_lds + wave * 16 * 32,
                       xrow + (size_t)(wave * 16) * CCH + k0, 32, 16, CCH);
    tdm_wait();
#else
    copy_b_tile(bt_lds, xrow + k0, CCH);
#endif
    __syncthreads();
    v16bf fa = frag_a(a_lds[wave], 32);
#pragma unroll
    for (int j = 0; j < 4; ++j)
      acc[j] = wmma_bf16(fa, frag_b(bt_lds + j * 16 * 32, 32), acc[j]);
    __syncthreads();
  }

  int nl = lane & 15, mb = (lane >> 4) << 3;
#pragma unroll
  for (int j = 0; j < 4; ++j) {
    int n = n0 + j * 16 + nl;
    union { uint4 q4; bf16 h[8]; } o;
#pragma unroll
    for (int i = 0; i < 8; ++i) {
      int d = d0 + wave * 16 + mb + i;
      float v = acc[j][i] + (bias ? bias[d] : 0.0f);
      o.h[i] = (bf16)v;
      if (out) out[((size_t)b * DO + d) * NPTS + n] = o.h[i];
    }
    if (outT)  // 8 consecutive d -> single 16B store
      *(uint4*)&outT[((size_t)b * NPTS + n) * DO + d0 + wave * 16 + mb] = o.q4;
  }
}

// ---------------------------------------------------------------------------
// K2: E = Q*Q^T Gram. Both operands are rows of Qbf (no transpose anywhere).
// sweep1: row maxima; sweep2: ST[m][n] = exp(E-max) with packed uint4 stores,
// per-row sums -> invrow = 1/rowsum.
// ---------------------------------------------------------------------------
__global__ void __launch_bounds__(128)
attn_softmax_kernel(const bf16* __restrict__ Qbf, bf16* __restrict__ ST,
                    float* __restrict__ invrow) {
  int b = blockIdx.y;
  int wave = threadIdx.x >> 5, lane = threadIdx.x & 31;
  int row0 = blockIdx.x * 64 + wave * 16;
  __shared__ __align__(16) bf16 a_lds[4][16 * DQK];
  __shared__ __align__(16) bf16 bt_lds[32 * DQK];

  const bf16* qb = Qbf + (size_t)b * NPTS * DQK;
#if USE_TDM
  tdm_load_tile_bf16(a_lds[wave], qb + (size_t)row0 * DQK, 16 * DQK, 1, 16 * DQK);
  tdm_wait();
#else
  {
    const uint4* s = (const uint4*)(qb + (size_t)row0 * DQK);
    uint4* d = (uint4*)a_lds[wave];
#pragma unroll
    for (int e = 0; e < 4; ++e) d[lane + 32 * e] = s[lane + 32 * e];
  }
#endif
  __syncthreads();

  int nl = lane & 15, mb = (lane >> 4) << 3;
  float rowmax[8];
#pragma unroll
  for (int i = 0; i < 8; ++i) rowmax[i] = -3.0e38f;

  // ---- sweep 1: row maxima only ----
  for (int mt = 0; mt < NPTS / 32; ++mt) {
    int col0 = mt * 32;
#if USE_TDM
    tdm_load_tile_bf16(bt_lds + wave * 8 * DQK, qb + (size_t)(col0 + wave * 8) * DQK,
                       8 * DQK, 1, 8 * DQK);
    tdm_wait();
#else
#pragma unroll
    for (int e = 0; e < 2; ++e) {
      int uid = threadIdx.x + 128 * e;   // 0..255
      ((uint4*)bt_lds)[uid] = ((const uint4*)(qb + (size_t)col0 * DQK))[uid];
    }
#endif
    __syncthreads();
#pragma unroll
    for (int j = 0; j < 2; ++j) {
      v8f acc = zero8();
#pragma unroll
      for (int kc = 0; kc < 2; ++kc)
        acc = wmma_bf16(frag_a(a_lds[wave] + kc * 32, DQK),
                        frag_b(bt_lds + j * 16 * DQK + kc * 32, DQK), acc);
#pragma unroll
      for (int i = 0; i < 8; ++i) rowmax[i] = fmaxf(rowmax[i], acc[i]);
    }
    __syncthreads();
  }
#pragma unroll
  for (int i = 0; i < 8; ++i) rowmax[i] = rmax16(rowmax[i]);

  // ---- sweep 2: S = exp(E - rowmax), packed transposed stores ----
  float rowsum[8];
#pragma unroll
  for (int i = 0; i < 8; ++i) rowsum[i] = 0.0f;
  for (int mt = 0; mt < NPTS / 32; ++mt) {
    int col0 = mt * 32;
#if USE_TDM
    tdm_load_tile_bf16(bt_lds + wave * 8 * DQK, qb + (size_t)(col0 + wave * 8) * DQK,
                       8 * DQK, 1, 8 * DQK);
    tdm_wait();
#else
#pragma unroll
    for (int e = 0; e < 2; ++e) {
      int uid = threadIdx.x + 128 * e;
      ((uint4*)bt_lds)[uid] = ((const uint4*)(qb + (size_t)col0 * DQK))[uid];
    }
#endif
    __syncthreads();
#pragma unroll
    for (int j = 0; j < 2; ++j) {
      v8f acc = zero8();
#pragma unroll
      for (int kc = 0; kc < 2; ++kc)
        acc = wmma_bf16(frag_a(a_lds[wave] + kc * 32, DQK),
                        frag_b(bt_lds + j * 16 * DQK + kc * 32, DQK), acc);
      int col = col0 + j * 16 + nl;
      union { uint4 q4; bf16 h[8]; } o;
#pragma unroll
      for (int i = 0; i < 8; ++i) {
        float s = __expf(acc[i] - rowmax[i]);
        rowsum[i] += s;
        o.h[i] = (bf16)s;
      }
      // lane's 8 values are 8 consecutive rows at fixed col -> ST[col][row0+mb..]
      *(uint4*)&ST[((size_t)b * NPTS + col) * NPTS + row0 + mb] = o.q4;
    }
    __syncthreads();
  }
#pragma unroll
  for (int i = 0; i < 8; ++i) rowsum[i] = rsum16(rowsum[i]);
  if (nl == 0) {
#pragma unroll
    for (int i = 0; i < 8; ++i)
      invrow[(size_t)b * NPTS + row0 + mb + i] = 1.0f / rowsum[i];
  }
}

// ---------------------------------------------------------------------------
// K3a: fold 1/rowsum into V (in place): V[b,c,n] *= invrow[b,n]
// ---------------------------------------------------------------------------
__global__ void vscale_kernel(bf16* __restrict__ V, const float* __restrict__ invrow) {
  size_t i = ((size_t)blockIdx.x * blockDim.x + threadIdx.x) * 2;
  int n = (int)(i & (NPTS - 1));
  int b = (int)(i / ((size_t)CCH * NPTS));
  const float* ir = invrow + (size_t)b * NPTS;
  unsigned int* p = (unsigned int*)(V + i);
  union { unsigned int u; bf16 h[2]; } w; w.u = *p;
  w.h[0] = (bf16)((float)w.h[0] * ir[n]);
  w.h[1] = (bf16)((float)w.h[1] * ir[n + 1]);
  *p = w.u;
}

// ---------------------------------------------------------------------------
// K3b: colsum[b,m] = sum_n ST[b,m,n] * invrow[b,n] — coalesced row reduction
// ---------------------------------------------------------------------------
__global__ void __launch_bounds__(256)
colsum_kernel(const bf16* __restrict__ ST, const float* __restrict__ invrow,
              float* __restrict__ colsum) {
  int m = blockIdx.x, b = blockIdx.y, tid = threadIdx.x;
  const bf16* row = ST + ((size_t)b * NPTS + m) * NPTS;
  const float* ir = invrow + (size_t)b * NPTS;
  __shared__ float red[256];
  float s = 0.0f;
  for (int i = tid; i < NPTS / 4; i += 256) {
    union { uint2 u; bf16 h[4]; } w; w.u = ((const uint2*)row)[i];
    float4 r4 = ((const float4*)ir)[i];
    s += (float)w.h[0] * r4.x + (float)w.h[1] * r4.y +
         (float)w.h[2] * r4.z + (float)w.h[3] * r4.w;
  }
  red[tid] = s;
  __syncthreads();
  for (int off = 128; off > 0; off >>= 1) {
    if (tid < off) red[tid] += red[tid + off];
    __syncthreads();
  }
  if (tid == 0) colsum[(size_t)b * NPTS + m] = red[0];
}

// ---------------------------------------------------------------------------
// K4: x_r = V'*S via ST tiles (row-major), scale 1/(1e-9+colsum),
// UT[n][c] = xT - x_r with packed uint4 load/store. A + B via TDM.
// ---------------------------------------------------------------------------
__global__ void __launch_bounds__(128)
apply_attn_kernel(const bf16* __restrict__ Vbf, const bf16* __restrict__ ST,
                  const float* __restrict__ colsum, const bf16* __restrict__ xT,
                  bf16* __restrict__ UT) {
  int b = blockIdx.z, c0 = blockIdx.y * 64, n0 = blockIdx.x * 64;
  int wave = threadIdx.x >> 5, lane = threadIdx.x & 31;
  __shared__ __align__(16) bf16 a_lds[4][16 * 32];
  __shared__ __align__(16) bf16 bt_lds[64 * 32];
  v8f acc[4];
#pragma unroll
  for (int j = 0; j < 4; ++j) acc[j] = zero8();

  const bf16* vb = Vbf + ((size_t)b * CCH + c0 + wave * 16) * NPTS;
  const bf16* st = ST + ((size_t)b * NPTS + n0) * NPTS;

  for (int k0 = 0; k0 < NPTS; k0 += 32) {
#if USE_TDM
    tdm_load_tile_bf16(a_lds[wave], vb + k0, 32, 16, NPTS);
    tdm_load_tile_bf16(bt_lds + wave * 16 * 32,
                       st + (size_t)(wave * 16) * NPTS + k0, 32, 16, NPTS);
    tdm_wait();
#else
    {
      uint4* d = (uint4*)a_lds[wave];
#pragma unroll
      for (int e = 0; e < 2; ++e) {
        int uid = lane + 32 * e;
        int r = uid >> 2, q = uid & 3;
        d[uid] = *(const uint4*)(vb + (size_t)r * NPTS + k0 + 8 * q);
      }
    }
    copy_b_tile(bt_lds, st + k0, NPTS);
#endif
    __syncthreads();
    v16bf fa = frag_a(a_lds[wave], 32);
#pragma unroll
    for (int j = 0; j < 4; ++j)
      acc[j] = wmma_bf16(fa, frag_b(bt_lds + j * 16 * 32, 32), acc[j]);
    __syncthreads();
  }

  int nl = lane & 15, mb = (lane >> 4) << 3;
#pragma unroll
  for (int j = 0; j < 4; ++j) {
    int n = n0 + j * 16 + nl;
    float inv = 1.0f / (1e-9f + colsum[(size_t)b * NPTS + n]);
    size_t base = ((size_t)b * NPTS + n) * CCH + c0 + wave * 16 + mb;
    union { uint4 q4; bf16 h[8]; } xv; xv.q4 = *(const uint4*)(xT + base);
    union { uint4 q4; bf16 h[8]; } uv;
#pragma unroll
    for (int i = 0; i < 8; ++i)
      uv.h[i] = (bf16)((float)xv.h[i] - acc[j][i] * inv);
    *(uint4*)(UT + base) = uv.q4;
  }
}

// ---------------------------------------------------------------------------
// K5: t = Wt*u + bt; B tiles from UT (row-major) via TDM
// ---------------------------------------------------------------------------
__global__ void __launch_bounds__(128)
trans_proj_kernel(const float* __restrict__ Wt, const float* __restrict__ bt,
                  const bf16* __restrict__ UT, float* __restrict__ T) {
  int b = blockIdx.z, d0 = blockIdx.y * 64, n0 = blockIdx.x * 64;
  int wave = threadIdx.x >> 5, lane = threadIdx.x & 31;
  __shared__ __align__(16) bf16 a_lds[4][16 * 32];
  __shared__ __align__(16) bf16 bt_lds[64 * 32];
  v8f acc[4];
#pragma unroll
  for (int j = 0; j < 4; ++j) acc[j] = zero8();

  const float* wrow = Wt + (size_t)(d0 + wave * 16) * CCH;
  const bf16* urow = UT + ((size_t)b * NPTS + n0) * CCH;

  for (int k0 = 0; k0 < CCH; k0 += 32) {
    float4 wa[4];
#pragma unroll
    for (int e = 0; e < 4; ++e) {
      int uid = lane + 32 * e;
      int r = uid >> 3, q = uid & 7;
      wa[e] = *(const float4*)(wrow + (size_t)r * CCH + k0 + 4 * q);
    }
#pragma unroll
    for (int e = 0; e < 4; ++e) {
      int uid = lane + 32 * e;
      int r = uid >> 3, q = uid & 7;
      const float* f = (const float*)&wa[e];
      union { uint2 u2; bf16 h[4]; } pk;
#pragma unroll
      for (int c = 0; c < 4; ++c) pk.h[c] = (bf16)f[c];
      *(uint2*)&a_lds[wave][r * 32 + 4 * q] = pk.u2;
    }
#if USE_TDM
    tdm_load_tile_bf16(bt_lds + wave * 16 * 32,
                       urow + (size_t)(wave * 16) * CCH + k0, 32, 16, CCH);
    tdm_wait();
#else
    copy_b_tile(bt_lds, urow + k0, CCH);
#endif
    __syncthreads();
    v16bf fa = frag_a(a_lds[wave], 32);
#pragma unroll
    for (int j = 0; j < 4; ++j)
      acc[j] = wmma_bf16(fa, frag_b(bt_lds + j * 16 * 32, 32), acc[j]);
    __syncthreads();
  }

  int nl = lane & 15, mb = (lane >> 4) << 3;
#pragma unroll
  for (int j = 0; j < 4; ++j) {
    int n = n0 + j * 16 + nl;
#pragma unroll
    for (int i = 0; i < 8; ++i) {
      int d = d0 + wave * 16 + mb + i;
      T[((size_t)b * CCH + d) * NPTS + n] = acc[j][i] + bt[d];
    }
  }
}

// ---------------------------------------------------------------------------
// K6: per-channel BN stats (deterministic tree reduction, one block/channel)
// ---------------------------------------------------------------------------
__global__ void bnstats_kernel(const float* __restrict__ T, float* __restrict__ bnsum,
                               float* __restrict__ bnsumsq) {
  int c = blockIdx.x, tid = threadIdx.x;
  __shared__ float ssum[256], ssq[256];
  float s = 0.0f, s2 = 0.0f;
  for (int b = 0; b < BATCH; ++b) {
    const float4* row = (const float4*)(T + ((size_t)b * CCH + c) * NPTS);
    for (int i = tid; i < NPTS / 4; i += 256) {
      float4 v = row[i];
      s  += v.x + v.y + v.z + v.w;
      s2 += v.x * v.x + v.y * v.y + v.z * v.z + v.w * v.w;
    }
  }
  ssum[tid] = s; ssq[tid] = s2;
  __syncthreads();
  for (int off = 128; off > 0; off >>= 1) {
    if (tid < off) { ssum[tid] += ssum[tid + off]; ssq[tid] += ssq[tid + off]; }
    __syncthreads();
  }
  if (tid == 0) { bnsum[c] = ssum[0]; bnsumsq[c] = ssq[0]; }
}

// ---------------------------------------------------------------------------
// K7: out = x + relu(BN(t)), float4 vectorized
// ---------------------------------------------------------------------------
__global__ void finalize_kernel(const float* __restrict__ x, const float* __restrict__ T,
                                const float* __restrict__ bnsum, const float* __restrict__ bnsumsq,
                                const float* __restrict__ gamma, const float* __restrict__ beta,
                                float* __restrict__ out) {
  size_t i4 = (size_t)blockIdx.x * blockDim.x + threadIdx.x;
  int c = (int)((i4 >> 10) & (CCH - 1));
  const float cnt = (float)(BATCH * NPTS);
  float mean = bnsum[c] / cnt;
  float var  = bnsumsq[c] / cnt - mean * mean;
  float rstd = rsqrtf(var + 1e-5f);
  float g = gamma[c] * rstd, be = beta[c] - mean * g;
  float4 t4 = ((const float4*)T)[i4];
  float4 x4 = ((const float4*)x)[i4];
  float4 r;
  r.x = x4.x + fmaxf(t4.x * g + be, 0.0f);
  r.y = x4.y + fmaxf(t4.y * g + be, 0.0f);
  r.z = x4.z + fmaxf(t4.z * g + be, 0.0f);
  r.w = x4.w + fmaxf(t4.w * g + be, 0.0f);
  ((float4*)out)[i4] = r;
}

extern "C" void kernel_launch(void* const* d_in, const int* in_sizes, int n_in,
                              void* d_out, int out_size, void* d_ws, size_t ws_size,
                              hipStream_t stream) {
  (void)in_sizes; (void)n_in; (void)out_size; (void)ws_size;
  const float* x     = (const float*)d_in[0];
  const float* Wqk   = (const float*)d_in[1];
  const float* Wv    = (const float*)d_in[2];
  const float* bv    = (const float*)d_in[3];
  const float* Wt    = (const float*)d_in[4];
  const float* btv   = (const float*)d_in[5];
  const float* gamma = (const float*)d_in[6];
  const float* beta  = (const float*)d_in[7];
  float* out = (float*)d_out;

  char* ws = (char*)d_ws;
  size_t off = 0;
  bf16* xT  = (bf16*)(ws + off); off += (size_t)BATCH * NPTS * CCH * 2;   // [B,N,C]
  bf16* Qbf = (bf16*)(ws + off); off += (size_t)BATCH * NPTS * DQK * 2;   // [B,N,64]
  bf16* Vbf = (bf16*)(ws + off); off += (size_t)BATCH * CCH * NPTS * 2;   // [B,C,N]
  bf16* UT  = (bf16*)(ws + off); off += (size_t)BATCH * NPTS * CCH * 2;   // [B,N,C]
  float* colsum = (float*)(ws + off); off += (size_t)BATCH * NPTS * 4;
  float* invrow = (float*)(ws + off); off += (size_t)BATCH * NPTS * 4;
  float* bnsum  = (float*)(ws + off); off += (size_t)CCH * 4;
  float* bnsq   = (float*)(ws + off); off += (size_t)CCH * 4;
  float* T      = (float*)(ws + off); off += (size_t)BATCH * CCH * NPTS * 4;
  bf16*  ST     = (bf16*)(ws + off);  off += (size_t)BATCH * NPTS * NPTS * 2;

  dim3 blk(128);
  xpose_kernel<<<dim3(NPTS / 64, CCH / 64, BATCH), dim3(256), 0, stream>>>(x, xT);
  proj_kernel<<<dim3(NPTS / 64, 1, BATCH), blk, 0, stream>>>(
      Wqk, nullptr, xT, nullptr, Qbf, DQK);
  proj_kernel<<<dim3(NPTS / 64, CCH / 64, BATCH), blk, 0, stream>>>(
      Wv, bv, xT, Vbf, nullptr, CCH);
  attn_softmax_kernel<<<dim3(NPTS / 64, BATCH), blk, 0, stream>>>(Qbf, ST, invrow);
  vscale_kernel<<<dim3((BATCH * CCH * NPTS / 2) / 256), dim3(256), 0, stream>>>(
      Vbf, invrow);
  colsum_kernel<<<dim3(NPTS, BATCH), dim3(256), 0, stream>>>(ST, invrow, colsum);
  apply_attn_kernel<<<dim3(NPTS / 64, CCH / 64, BATCH), blk, 0, stream>>>(
      Vbf, ST, colsum, xT, UT);
  trans_proj_kernel<<<dim3(NPTS / 64, CCH / 64, BATCH), blk, 0, stream>>>(
      Wt, btv, UT, T);
  bnstats_kernel<<<dim3(CCH), dim3(256), 0, stream>>>(T, bnsum, bnsq);
  finalize_kernel<<<dim3((BATCH * CCH * NPTS / 4) / 256), dim3(256), 0, stream>>>(
      x, T, bnsum, bnsq, gamma, beta, out);
}
